// GCNClassifier_33861522161794
// MI455X (gfx1250) — matrix operand order repaired and checked by reading, not compiled, not run on hardware
//
#include <hip/hip_runtime.h>
#include <hip/hip_bf16.h>
#include <math.h>

#define N_NODES 100000
#define N_EDGES 300000
#define FDIM    256

typedef __attribute__((ext_vector_type(16))) __bf16 v16bf;
typedef __attribute__((ext_vector_type(8)))  float  v8f;

union ABop { v16bf v; uint4 q[2]; };

// ---------- bf16 split helpers (round-to-nearest-even) ----------
__device__ __forceinline__ unsigned short f2bf_bits(float f) {
    union { float f; unsigned u; } v; v.f = f;
    unsigned r = v.u + 0x7FFFu + ((v.u >> 16) & 1u);   // RNE
    return (unsigned short)(r >> 16);
}
__device__ __forceinline__ float bfbits2f(unsigned short h) {
    union { unsigned u; float f; } v; v.u = ((unsigned)h) << 16; return v.f;
}

// ---------------- degree / normalization ----------------
__global__ void gcn_deg_init(float* deg, int n) {
    int i = blockIdx.x * blockDim.x + threadIdx.x;
    if (i < n) deg[i] = 1.0f;                 // self loop
}
__global__ void gcn_deg_edges(const int* __restrict__ ei, float* deg, int e) {
    int i = blockIdx.x * blockDim.x + threadIdx.x;
    if (i < e) atomicAdd(&deg[ei[e + i]], 1.0f);   // dst row of edge_index
}
__global__ void gcn_dinv(float* deg, int n) {
    int i = blockIdx.x * blockDim.x + threadIdx.x;
    if (i < n) deg[i] = rsqrtf(deg[i]);       // deg >= 1 always (self loop)
}

// ---------------- one-time W split into per-lane WMMA-B layout ----------------
// out index o = ((nt*8 + kt)*32 + lane)*16 + j  maps to
//   column n = nt*16 + (lane&15),  K = kt*32 + (lane>>4)*8 + (j&7) + (j>>3)*16
__global__ void gcn_prep_w(const float* __restrict__ W,
                           unsigned short* __restrict__ w_hi,
                           unsigned short* __restrict__ w_lo) {
    int o    = blockIdx.x * blockDim.x + threadIdx.x;   // 0 .. 65535
    int j    = o & 15;
    int lane = (o >> 4) & 31;
    int ktnt = o >> 9;
    int kt   = ktnt & 7;
    int nt   = ktnt >> 3;
    int n    = nt * 16 + (lane & 15);
    int k    = kt * 32 + (lane >> 4) * 8 + (j & 7) + (j >> 3) * 16;
    float f  = W[(size_t)k * FDIM + n];
    unsigned short hb = f2bf_bits(f);
    w_hi[o] = hb;
    w_lo[o] = f2bf_bits(f - bfbits2f(hb));
}

// ---------------- h = x @ W_conv via split-bf16 WMMA ----------------
// grid.x = N/16 row tiles; block = 256 threads (8 waves); wave w owns N-tiles {2w, 2w+1}
#define ROWS 264   // 256 bf16 + 8 pad (528 B row stride -> conflict-free)
__global__ void __launch_bounds__(256)
gcn_gemm_wmma(const float* __restrict__ x,
              const unsigned short* __restrict__ w_hi,
              const unsigned short* __restrict__ w_lo,
              float* __restrict__ h) {
    __shared__ unsigned short a_hi_s[16 * ROWS];
    __shared__ unsigned short a_lo_s[16 * ROWS];

    const int tid   = threadIdx.x;
    const int mrow0 = blockIdx.x * 16;

    // stage + split 16x256 slab of x into LDS bf16 hi/lo (coalesced, one row per iter)
    #pragma unroll
    for (int it = 0; it < 16; ++it) {
        float f = x[(size_t)(mrow0 + it) * FDIM + tid];
        unsigned short hb = f2bf_bits(f);
        a_hi_s[it * ROWS + tid] = hb;
        a_lo_s[it * ROWS + tid] = f2bf_bits(f - bfbits2f(hb));
    }
    __syncthreads();

    const int wave = tid >> 5;
    const int lane = tid & 31;
    const int ml   = lane & 15;               // A row / B,C,D column-in-tile
    const int g    = lane >> 4;               // half-wave group
    const int nt0  = wave * 2, nt1 = nt0 + 1;

    const uint4* wh = (const uint4*)w_hi;     // [(nt*8+kt)*32+lane] -> 2 x uint4
    const uint4* wl = (const uint4*)w_lo;

    v8f acc0 = {}, acc1 = {};
    #pragma unroll
    for (int kt = 0; kt < FDIM / 32; ++kt) {
        // A operands: two contiguous 8-bf16 runs per lane, both 16B aligned
        const int aoff = ml * ROWS + kt * 32 + g * 8;
        ABop a_hi, a_lo;
        a_hi.q[0] = *(const uint4*)&a_hi_s[aoff];
        a_hi.q[1] = *(const uint4*)&a_hi_s[aoff + 16];
        a_lo.q[0] = *(const uint4*)&a_lo_s[aoff];
        a_lo.q[1] = *(const uint4*)&a_lo_s[aoff + 16];

        // B operands: 32 contiguous bytes per lane from pre-swizzled W
        const size_t b0 = ((size_t)(nt0 * 8 + kt) * 32 + lane) * 2;
        const size_t b1 = ((size_t)(nt1 * 8 + kt) * 32 + lane) * 2;
        ABop bh0, bl0, bh1, bl1;
        bh0.q[0] = wh[b0]; bh0.q[1] = wh[b0 + 1];
        bl0.q[0] = wl[b0]; bl0.q[1] = wl[b0 + 1];
        bh1.q[0] = wh[b1]; bh1.q[1] = wh[b1 + 1];
        bl1.q[0] = wl[b1]; bl1.q[1] = wl[b1 + 1];

        // split-bf16: hi*hi + hi*lo + lo*hi  (~fp32 accuracy)
        acc0 = __builtin_amdgcn_wmma_f32_16x16x32_bf16(false, a_hi.v, false, bh0.v,
                                                       (short)0, acc0, false, false);
        acc0 = __builtin_amdgcn_wmma_f32_16x16x32_bf16(false, a_hi.v, false, bl0.v,
                                                       (short)0, acc0, false, false);
        acc0 = __builtin_amdgcn_wmma_f32_16x16x32_bf16(false, a_lo.v, false, bh0.v,
                                                       (short)0, acc0, false, false);
        acc1 = __builtin_amdgcn_wmma_f32_16x16x32_bf16(false, a_hi.v, false, bh1.v,
                                                       (short)0, acc1, false, false);
        acc1 = __builtin_amdgcn_wmma_f32_16x16x32_bf16(false, a_hi.v, false, bl1.v,
                                                       (short)0, acc1, false, false);
        acc1 = __builtin_amdgcn_wmma_f32_16x16x32_bf16(false, a_lo.v, false, bh1.v,
                                                       (short)0, acc1, false, false);
    }

    // D: lane l element r -> (M = r + 8*g, N = nt*16 + ml)
    #pragma unroll
    for (int r = 0; r < 8; ++r) {
        h[(size_t)(mrow0 + r + 8 * g) * FDIM + nt0 * 16 + ml] = acc0[r];
        h[(size_t)(mrow0 + r + 8 * g) * FDIM + nt1 * 16 + ml] = acc1[r];
    }
}

// ---------------- agg init with self-loop message: agg = h * dinv^2 ----------------
__global__ void gcn_selfloop(const float* __restrict__ h, const float* __restrict__ dinv,
                             float* __restrict__ agg) {
    int idx  = blockIdx.x * blockDim.x + threadIdx.x;   // over N*F (exact multiple)
    int node = idx >> 8;                                 // /256
    float d  = dinv[node];
    agg[idx] = h[idx] * d * d;
}

// ---------------- edge scatter: agg[dst] += h[src] * dinv[src]*dinv[dst] ----------------
__global__ void gcn_scatter(const float* __restrict__ h, const float* __restrict__ dinv,
                            const int* __restrict__ ei, float* __restrict__ agg, int e) {
    int edge = blockIdx.x;                               // one block per edge
    int feat = threadIdx.x;                              // 256 features
    int src  = ei[edge];
    int dst  = ei[e + edge];
    float nrm = dinv[src] * dinv[dst];
    atomicAdd(&agg[(size_t)dst * FDIM + feat], h[(size_t)src * FDIM + feat] * nrm);
}

// ---------------- head: sigmoid(relu(agg + b_conv) @ W_lin + b_lin) ----------------
// one wave (32 lanes) per node; 8 nodes per 256-thread block
__global__ void __launch_bounds__(256)
gcn_head(const float* __restrict__ agg, const float* __restrict__ b_conv,
         const float* __restrict__ W_lin, const float* __restrict__ b_lin,
         float* __restrict__ out) {
    int wave = threadIdx.x >> 5;
    int lane = threadIdx.x & 31;
    int node = blockIdx.x * 8 + wave;

    float s = 0.0f;
    #pragma unroll
    for (int r = 0; r < 8; ++r) {
        int f = r * 32 + lane;
        float v = agg[(size_t)node * FDIM + f] + b_conv[f];
        v = fmaxf(v, 0.0f);
        s += v * W_lin[f];
    }
    #pragma unroll
    for (int off = 16; off > 0; off >>= 1)
        s += __shfl_down(s, off, 32);
    if (lane == 0) {
        float z = s + b_lin[0];
        out[node] = 1.0f / (1.0f + expf(-z));
    }
}

extern "C" void kernel_launch(void* const* d_in, const int* in_sizes, int n_in,
                              void* d_out, int out_size, void* d_ws, size_t ws_size,
                              hipStream_t stream) {
    const float* x      = (const float*)d_in[0];
    const int*   ei     = (const int*)  d_in[1];   // [2, E]: src then dst
    const float* W_conv = (const float*)d_in[2];
    const float* b_conv = (const float*)d_in[3];
    const float* W_lin  = (const float*)d_in[4];
    const float* b_lin  = (const float*)d_in[5];
    float* out = (float*)d_out;

    // workspace: h [N*F] | agg [N*F] | dinv [N] | w_hi [F*F u16] | w_lo [F*F u16]
    float* h    = (float*)d_ws;
    float* agg  = h   + (size_t)N_NODES * FDIM;
    float* dinv = agg + (size_t)N_NODES * FDIM;
    unsigned short* w_hi = (unsigned short*)(dinv + N_NODES);
    unsigned short* w_lo = w_hi + (size_t)FDIM * FDIM;

    const int N = N_NODES, E = N_EDGES;

    gcn_deg_init <<<(N + 255) / 256, 256, 0, stream>>>(dinv, N);
    gcn_deg_edges<<<(E + 255) / 256, 256, 0, stream>>>(ei, dinv, E);
    gcn_dinv     <<<(N + 255) / 256, 256, 0, stream>>>(dinv, N);

    gcn_prep_w   <<<(FDIM * FDIM) / 256, 256, 0, stream>>>(W_conv, w_hi, w_lo);
    gcn_gemm_wmma<<<N / 16, 256, 0, stream>>>(x, w_hi, w_lo, h);

    gcn_selfloop <<<(size_t)N * FDIM / 256, 256, 0, stream>>>(h, dinv, agg);
    gcn_scatter  <<<E, 256, 0, stream>>>(h, dinv, ei, agg, E);

    gcn_head     <<<N / 8, 256, 0, stream>>>(agg, b_conv, W_lin, b_lin, out);
}
